// MultiHeadAttention1D_10015863734574
// MI455X (gfx1250) — matrix operand order repaired and checked by reading, not compiled, not run on hardware
//
#include <hip/hip_runtime.h>

// ---------------------------------------------------------------------------
// MultiHeadAttention1D for MI455X (gfx1250).
// B=4, S=2048, HIDDEN=1024, H=16, D=64. bf16 WMMA (fp32 accum) everywhere,
// TDM (tensor_load_to_lds) staged K/V tiles in flash attention.
// ---------------------------------------------------------------------------

typedef __bf16 bf16;
typedef __attribute__((ext_vector_type(16))) __bf16        v16bf;
typedef __attribute__((ext_vector_type(8)))  __bf16        v8bf;
typedef __attribute__((ext_vector_type(8)))  float         v8f;
typedef __attribute__((ext_vector_type(4)))  unsigned int  v4u;
typedef __attribute__((ext_vector_type(8)))  int           v8i;
typedef __attribute__((ext_vector_type(4)))  int           v4i;

#define BATCH    4
#define SEQ      2048
#define HIDDEN   1024
#define NHEADS   16
#define HDIM     64
#define ROWS     (BATCH * SEQ)          // 8192
#define MASK_BIAS (-10000.0f)

static __device__ inline v8f wmma_bf16(v16bf a, v16bf b, v8f c) {
  return __builtin_amdgcn_wmma_f32_16x16x32_bf16(false, a, false, b, (short)0, c,
                                                 false, false);
}

// A-fragment (16x32 bf16, M x K). `rowk` points at element [row=lane&15][kbase].
// Lanes 0-15 hold K=0..15, lanes 16-31 hold K=16..31 (contiguous, matches B).
static __device__ inline v16bf load_a_frag(const bf16* rowk, int lane) {
  const int kb = (lane & 16) ? 16 : 0;
  v8bf lo = *reinterpret_cast<const v8bf*>(rowk + kb);
  v8bf hi = *reinterpret_cast<const v8bf*>(rowk + kb + 8);
  v16bf r;
#pragma unroll
  for (int i = 0; i < 8; ++i) { r[i] = lo[i]; r[i + 8] = hi[i]; }
  return r;
}

// ---------------------------------------------------------------------------
// Tensor Data Mover: 2D tile (tile_w x tile_h elements, 2-byte) -> LDS.
// D# layout per CDNA5 ISA ch.8: group0 = {flags, lds_addr, global_addr, type},
// group1 = {data_size, tensor dims, tile dims, dim0 stride}.
// ---------------------------------------------------------------------------
#if defined(__has_builtin)
#if __has_builtin(__builtin_amdgcn_tensor_load_to_lds)
#define HAS_TDM 1
#endif
#endif
#ifndef HAS_TDM
#define HAS_TDM 0
#endif

static __device__ inline void tdm_load_2d(const bf16* gaddr, unsigned lds_off,
                                          unsigned tile_w, unsigned tile_h,
                                          unsigned tensor_w, unsigned tensor_h,
                                          unsigned long long row_stride_elems) {
#if HAS_TDM
  unsigned long long ga = (unsigned long long)(uintptr_t)gaddr;
  v4u g0;
  g0[0] = 1u;                                              // count=1 valid D#
  g0[1] = lds_off;                                         // lds_addr (bytes)
  g0[2] = (unsigned)ga;                                    // global_addr[31:0]
  g0[3] = (unsigned)((ga >> 32) & 0x1FFFFFFu) | (2u << 30); // addr[56:32] | type=2
  v8i g1;
  g1[0] = (int)(1u << 16);                                 // data_size=1 (2 bytes)
  g1[1] = (int)((tensor_w & 0xFFFFu) << 16);               // tensor_dim0[15:0]
  g1[2] = (int)(((tensor_w >> 16) & 0xFFFFu) | ((tensor_h & 0xFFFFu) << 16));
  g1[3] = (int)(((tensor_h >> 16) & 0xFFFFu) | ((tile_w & 0xFFFFu) << 16));
  g1[4] = (int)(tile_h & 0xFFFFu);                         // tile_dim1 (tile_dim2=0)
  g1[5] = (int)(unsigned)row_stride_elems;                 // tensor_dim0_stride[31:0]
  g1[6] = (int)(unsigned)((row_stride_elems >> 32) & 0xFFFFu);
  g1[7] = 0;
  v4i z4 = {0, 0, 0, 0};
#if __clang_major__ >= 23
  v8i z8 = {0, 0, 0, 0, 0, 0, 0, 0};
  __builtin_amdgcn_tensor_load_to_lds(g0, g1, z4, z4, z8, 0);
#else
  __builtin_amdgcn_tensor_load_to_lds(g0, g1, z4, z4, 0);
#endif
#else
  (void)gaddr; (void)lds_off; (void)tile_w; (void)tile_h;
  (void)tensor_w; (void)tensor_h; (void)row_stride_elems;
#endif
}

static __device__ inline void wait_tensorcnt0() {
#if defined(__has_builtin)
#if __has_builtin(__builtin_amdgcn_s_wait_tensorcnt)
  __builtin_amdgcn_s_wait_tensorcnt((short)0);
  return;
#endif
#endif
  asm volatile("s_wait_tensorcnt 0" ::: "memory");
}

// ---------------------------------------------------------------------------
// Conversions
// ---------------------------------------------------------------------------
__global__ __launch_bounds__(256) void conv_f32_to_bf16(const float* __restrict__ in,
                                                        bf16* __restrict__ out, int n) {
  int i = blockIdx.x * 256 + threadIdx.x;
  if (i < n) out[i] = (bf16)in[i];
}

// w [K][N] fp32 row-major -> wT [N][K] bf16
__global__ __launch_bounds__(256) void conv_transpose_bf16(const float* __restrict__ w,
                                                           bf16* __restrict__ wT,
                                                           int K, int N) {
  int i = blockIdx.x * 256 + threadIdx.x;
  if (i < K * N) {
    int n = i / K, k = i - n * K;
    wT[i] = (bf16)w[(size_t)k * N + n];
  }
}

// ---------------------------------------------------------------------------
// QKV GEMM: X[8192,1024] @ W[1024,3072] + bias -> Q/K/Vt bf16.
// 32M x 64N register tile per wave: 8 accumulators, A reused 4x, B reused 2x.
// Fused chunk order is k, v, q (key first!).
// ---------------------------------------------------------------------------
__global__ __launch_bounds__(256) void qkv_gemm(const bf16* __restrict__ X,
                                                const bf16* __restrict__ WT,    // [3072][1024]
                                                const float* __restrict__ bias, // [3072]
                                                bf16* __restrict__ Qb,   // [B,H,S,D]
                                                bf16* __restrict__ Kb,   // [B,H,S,D]
                                                bf16* __restrict__ Vtb)  // [B,H,D,S]
{
  const int lane = threadIdx.x & 31;
  const int gid  = blockIdx.x * 8 + (threadIdx.x >> 5);
  const int tm = gid & 255;        // 256 M tiles of 32
  const int tn = gid >> 8;         // 48 N tiles of 64
  const int m0 = tm * 32, n0 = tn * 64;
  const int r = lane & 15;
  const int halfk = (lane & 16) ? 16 : 0;
  const int mhalf = (lane & 16) ? 8 : 0;

  const bf16* arow0 = X + (size_t)(m0 + r) * HIDDEN;
  const bf16* arow1 = X + (size_t)(m0 + 16 + r) * HIDDEN;
  const bf16* brow[4];
#pragma unroll
  for (int ni = 0; ni < 4; ++ni)
    brow[ni] = WT + (size_t)(n0 + ni * 16 + r) * HIDDEN + halfk;

  v8f c[2][4];
#pragma unroll
  for (int mi = 0; mi < 2; ++mi)
#pragma unroll
    for (int ni = 0; ni < 4; ++ni)
#pragma unroll
      for (int v = 0; v < 8; ++v) c[mi][ni][v] = 0.f;

  for (int kb = 0; kb < HIDDEN; kb += 32) {
    v16bf a0 = load_a_frag(arow0 + kb, lane);
    v16bf a1 = load_a_frag(arow1 + kb, lane);
#pragma unroll
    for (int ni = 0; ni < 4; ++ni) {
      v16bf b = *reinterpret_cast<const v16bf*>(brow[ni] + kb);
      c[0][ni] = wmma_bf16(a0, b, c[0][ni]);
      c[1][ni] = wmma_bf16(a1, b, c[1][ni]);
    }
  }

  // Epilogue. C/D layout: lanes 0-15 -> M=v, N=lane; lanes 16-31 -> M=v+8, N=lane-16
#pragma unroll
  for (int ni = 0; ni < 4; ++ni) {
    const int col = n0 + ni * 16 + r;       // 0..3071
    const float bc = bias[col];
    const int which = col >> 10;            // 0=k, 1=v, 2=q
    const int cc = col & 1023;
    const int h = cc >> 6, d = cc & 63;
#pragma unroll
    for (int mi = 0; mi < 2; ++mi) {
#pragma unroll
      for (int v = 0; v < 8; ++v) {
        const int m = m0 + mi * 16 + v + mhalf;
        const int bi = m >> 11, s = m & 2047;
        const float val = c[mi][ni][v] + bc;
        const size_t bh = (size_t)bi * NHEADS + h;
        if (which == 0) {
          Kb[(bh * SEQ + s) * HDIM + d] = (bf16)val;
        } else if (which == 1) {
          Vtb[(bh * HDIM + d) * SEQ + s] = (bf16)val;
        } else {
          Qb[(bh * SEQ + s) * HDIM + d] = (bf16)(val * 0.125f);  // D^-0.5
        }
      }
    }
  }
}

// ---------------------------------------------------------------------------
// Flash attention. One block per (b,h) x 128 query rows (wave w -> 16 rows).
// K (32x64) and Vt (64x32) tiles staged in LDS by the Tensor Data Mover,
// double buffered; one barrier per KV step; all 8 waves consume from LDS.
// ---------------------------------------------------------------------------
__global__ __launch_bounds__(256) void flash_attn(const bf16* __restrict__ Qb,
                                                  const bf16* __restrict__ Kb,
                                                  const bf16* __restrict__ Vtb,
                                                  bf16* __restrict__ Ctx)   // [B,S,HIDDEN]
{
  __shared__ __attribute__((aligned(64))) bf16 kbuf[2][32 * HDIM];  // 8 KB
  __shared__ __attribute__((aligned(64))) bf16 vbuf[2][HDIM * 32];  // 8 KB
  __shared__ __attribute__((aligned(32))) bf16 plds[8][16 * 32];    // 8 KB

  const int lane = threadIdx.x & 31;
  const int wv   = threadIdx.x >> 5;
  const int qblk = blockIdx.x & 15;       // 16 q-blocks of 128 rows
  const int bh   = blockIdx.x >> 4;       // b*16 + h
  const int b    = bh >> 4, h = bh & 15;

  const size_t bhoff = (size_t)bh * SEQ * HDIM;
  const bf16* Qp = Qb + bhoff;            // [SEQ][HDIM]
  const bf16* Kp = Kb + bhoff;            // [SEQ][HDIM]
  const bf16* Vt = Vtb + bhoff;           // [HDIM][SEQ]

  const int qb0   = qblk * 128;
  const int qbase = qb0 + wv * 16;
  const int r = lane & 15;
  const int halfk = (lane & 16) ? 16 : 0;
  const int mhalf = (lane & 16) ? 8 : 0;

  // Q fragments for K(head-dim) 0..31 and 32..63
  const bf16* qrow = Qp + (size_t)(qbase + r) * HDIM;
  const v16bf qa0 = load_a_frag(qrow, lane);
  const v16bf qa1 = load_a_frag(qrow + 32, lane);

  float mrow[8], lrow[8];
  v8f acc[4];
#pragma unroll
  for (int v = 0; v < 8; ++v) { mrow[v] = -1e30f; lrow[v] = 0.f; }
#pragma unroll
  for (int dt = 0; dt < 4; ++dt)
#pragma unroll
    for (int v = 0; v < 8; ++v) acc[dt][v] = 0.f;

  const int nsteps = (qb0 + 128) / 32;    // block-uniform causal KV span

#if HAS_TDM
  if (wv == 0) {   // preload buffer 0 (kv0 = 0)
    tdm_load_2d(Kp, (unsigned)(uintptr_t)&kbuf[0][0], HDIM, 32, HDIM, SEQ, HDIM);
    tdm_load_2d(Vt, (unsigned)(uintptr_t)&vbuf[0][0], 32, HDIM, SEQ, HDIM, SEQ);
  }
#endif

  for (int i = 0; i < nsteps; ++i) {
    const int kv0 = i * 32;
    const int buf = i & 1;

#if HAS_TDM
    if (wv == 0) wait_tensorcnt0();       // tile for this step landed in LDS
    __syncthreads();                      // release consumers / retire prev reads
    if (wv == 0 && i + 1 < nsteps) {      // DMA next tile while we compute
      const int kn = kv0 + 32;
      tdm_load_2d(Kp + (size_t)kn * HDIM, (unsigned)(uintptr_t)&kbuf[buf ^ 1][0],
                  HDIM, 32, HDIM, SEQ, HDIM);
      tdm_load_2d(Vt + kn, (unsigned)(uintptr_t)&vbuf[buf ^ 1][0],
                  32, HDIM, SEQ, HDIM, SEQ);
    }
#else
    __syncthreads();
    // cooperative fallback: 256 threads copy K (2048) + Vt (2048) elements
    for (int e = threadIdx.x; e < 32 * HDIM; e += 256) {
      kbuf[buf][e] = Kp[(size_t)(kv0 + (e >> 6)) * HDIM + (e & 63)];
      vbuf[buf][e] = Vt[(size_t)(e >> 5) * SEQ + kv0 + (e & 31)];
    }
    __syncthreads();
#endif

    if (kv0 <= qbase + 15) {              // causal: this wave still has work
      const bf16* kl = &kbuf[buf][0];
      const bf16* vl = &vbuf[buf][0];

      // ---- scores: S = Q(16x64) * K^T(64x32) -> two 16x16 accumulators
      v8f s0 = {0.f, 0.f, 0.f, 0.f, 0.f, 0.f, 0.f, 0.f};
      v8f s1 = s0;
      const bf16* krow0 = kl + (size_t)r * HDIM;           // kv rows 0..15
      const bf16* krow1 = kl + (size_t)(16 + r) * HDIM;    // kv rows 16..31
      s0 = wmma_bf16(qa0, *reinterpret_cast<const v16bf*>(krow0 + halfk), s0);
      s0 = wmma_bf16(qa1, *reinterpret_cast<const v16bf*>(krow0 + 32 + halfk), s0);
      s1 = wmma_bf16(qa0, *reinterpret_cast<const v16bf*>(krow1 + halfk), s1);
      s1 = wmma_bf16(qa1, *reinterpret_cast<const v16bf*>(krow1 + 32 + halfk), s1);

      // ---- causal mask + online softmax (row = (v, half-wave))
#pragma unroll
      for (int v = 0; v < 8; ++v) {
        const int M = v + mhalf;
        const int qrowi = qbase + M;
        const int c0 = kv0 + r, c1 = c0 + 16;
        float x0 = (c0 <= qrowi) ? s0[v] : MASK_BIAS;
        float x1 = (c1 <= qrowi) ? s1[v] : MASK_BIAS;

        float mx = fmaxf(x0, x1);
#pragma unroll
        for (int off = 1; off < 16; off <<= 1)
          mx = fmaxf(mx, __shfl_xor(mx, off, 32));
        const float mnew = fmaxf(mrow[v], mx);
        const float sc = __expf(mrow[v] - mnew);
        const float p0 = __expf(x0 - mnew);
        const float p1 = __expf(x1 - mnew);
        float rs = p0 + p1;
#pragma unroll
        for (int off = 1; off < 16; off <<= 1)
          rs += __shfl_xor(rs, off, 32);
        lrow[v] = lrow[v] * sc + rs;
        mrow[v] = mnew;
#pragma unroll
        for (int dt = 0; dt < 4; ++dt) acc[dt][v] *= sc;

        plds[wv][M * 32 + r]      = (bf16)p0;
        plds[wv][M * 32 + 16 + r] = (bf16)p1;
      }

      // LDS ops are wave-in-order; wait for store data before re-reading.
      asm volatile("s_wait_dscnt 0" ::: "memory");

      // ---- P tile (16x32) as A fragment; PV: acc += P * V(32x64)
      const v16bf pa = load_a_frag(&plds[wv][r * 32], lane);
#pragma unroll
      for (int dt = 0; dt < 4; ++dt) {
        const bf16* vrow = vl + (size_t)(dt * 16 + r) * 32 + halfk;
        acc[dt] = wmma_bf16(pa, *reinterpret_cast<const v16bf*>(vrow), acc[dt]);
      }
    }
  }

  // ---- normalize and write ctx[b, s, h*64 + d] as bf16
#pragma unroll
  for (int v = 0; v < 8; ++v) {
    const int M = v + mhalf;
    const int s = qbase + M;
    const float rinv = 1.0f / lrow[v];
#pragma unroll
    for (int dt = 0; dt < 4; ++dt) {
      const int col = h * HDIM + dt * 16 + r;
      Ctx[((size_t)b * SEQ + s) * HIDDEN + col] = (bf16)(acc[dt][v] * rinv);
    }
  }
}

// ---------------------------------------------------------------------------
// Output GEMM: ctx[8192,1024] @ w_dense[1024,1024] + bias -> fp32 out.
// Same 32M x 64N register tiling.
// ---------------------------------------------------------------------------
__global__ __launch_bounds__(256) void dense_gemm(const bf16* __restrict__ X,
                                                  const bf16* __restrict__ WT,  // [1024][1024]
                                                  const float* __restrict__ bias,
                                                  float* __restrict__ out) {
  const int lane = threadIdx.x & 31;
  const int gid  = blockIdx.x * 8 + (threadIdx.x >> 5);
  const int tm = gid & 255;       // 256 M tiles of 32
  const int tn = gid >> 8;        // 16 N tiles of 64
  const int m0 = tm * 32, n0 = tn * 64;
  const int r = lane & 15;
  const int halfk = (lane & 16) ? 16 : 0;
  const int mhalf = (lane & 16) ? 8 : 0;

  const bf16* arow0 = X + (size_t)(m0 + r) * HIDDEN;
  const bf16* arow1 = X + (size_t)(m0 + 16 + r) * HIDDEN;
  const bf16* brow[4];
#pragma unroll
  for (int ni = 0; ni < 4; ++ni)
    brow[ni] = WT + (size_t)(n0 + ni * 16 + r) * HIDDEN + halfk;

  v8f c[2][4];
#pragma unroll
  for (int mi = 0; mi < 2; ++mi)
#pragma unroll
    for (int ni = 0; ni < 4; ++ni)
#pragma unroll
      for (int v = 0; v < 8; ++v) c[mi][ni][v] = 0.f;

  for (int kb = 0; kb < HIDDEN; kb += 32) {
    v16bf a0 = load_a_frag(arow0 + kb, lane);
    v16bf a1 = load_a_frag(arow1 + kb, lane);
#pragma unroll
    for (int ni = 0; ni < 4; ++ni) {
      v16bf b = *reinterpret_cast<const v16bf*>(brow[ni] + kb);
      c[0][ni] = wmma_bf16(a0, b, c[0][ni]);
      c[1][ni] = wmma_bf16(a1, b, c[1][ni]);
    }
  }

#pragma unroll
  for (int ni = 0; ni < 4; ++ni) {
    const int col = n0 + ni * 16 + r;
    const float bc = bias[col];
#pragma unroll
    for (int mi = 0; mi < 2; ++mi) {
#pragma unroll
      for (int v = 0; v < 8; ++v) {
        const int m = m0 + mi * 16 + v + mhalf;
        out[(size_t)m * HIDDEN + col] = c[mi][ni][v] + bc;
      }
    }
  }
}

// ---------------------------------------------------------------------------
// Launch
// ---------------------------------------------------------------------------
extern "C" void kernel_launch(void* const* d_in, const int* in_sizes, int n_in,
                              void* d_out, int out_size, void* d_ws, size_t ws_size,
                              hipStream_t stream) {
  const float* hidden  = (const float*)d_in[0];  // [4,2048,1024]
  const float* w_qkv   = (const float*)d_in[1];  // [1024,3072]
  const float* b_qkv   = (const float*)d_in[2];  // [3072]
  const float* w_dense = (const float*)d_in[3];  // [1024,1024]
  const float* b_dense = (const float*)d_in[4];  // [1024]
  float* out = (float*)d_out;                    // [4,2048,1024]

  // Workspace carve-out (all 256B aligned): ~88 MiB total.
  char* ws = (char*)d_ws;
  size_t off = 0;
  auto take = [&](size_t bytes) {
    void* p = ws + off;
    off += (bytes + 255) & ~(size_t)255;
    return p;
  };
  bf16* Xbf   = (bf16*)take((size_t)ROWS * HIDDEN * 2);        // 16 MiB
  bf16* WqkvT = (bf16*)take((size_t)3 * HIDDEN * HIDDEN * 2);  //  6 MiB
  bf16* WdT   = (bf16*)take((size_t)HIDDEN * HIDDEN * 2);      //  2 MiB
  bf16* Qb    = (bf16*)take((size_t)ROWS * HIDDEN * 2);        // 16 MiB
  bf16* Kb    = (bf16*)take((size_t)ROWS * HIDDEN * 2);        // 16 MiB
  bf16* Vtb   = (bf16*)take((size_t)ROWS * HIDDEN * 2);        // 16 MiB
  bf16* Ctx   = (bf16*)take((size_t)ROWS * HIDDEN * 2);        // 16 MiB
  (void)ws_size; (void)in_sizes; (void)n_in; (void)out_size;

  // 1) conversions
  conv_f32_to_bf16<<<(ROWS * HIDDEN) / 256, 256, 0, stream>>>(hidden, Xbf, ROWS * HIDDEN);
  conv_transpose_bf16<<<(3 * HIDDEN * HIDDEN) / 256, 256, 0, stream>>>(w_qkv, WqkvT, HIDDEN, 3 * HIDDEN);
  conv_transpose_bf16<<<(HIDDEN * HIDDEN) / 256, 256, 0, stream>>>(w_dense, WdT, HIDDEN, HIDDEN);

  // 2) fused QKV projection: 256 (M/32) x 48 (N/64) waves, 8 waves/block
  qkv_gemm<<<(256 * 48) / 8, 256, 0, stream>>>(Xbf, WqkvT, b_qkv, Qb, Kb, Vtb);

  // 3) causal flash attention: block = (b,h) x 128 q rows -> 64*16 = 1024 blocks
  flash_attn<<<1024, 256, 0, stream>>>(Qb, Kb, Vtb, Ctx);

  // 4) output dense: 256 (M/32) x 16 (N/64) waves
  dense_gemm<<<(256 * 16) / 8, 256, 0, stream>>>(Ctx, WdT, b_dense, out);
}
